// BANLayer_13975823581592
// MI455X (gfx1250) — matrix operand order repaired and checked by reading, not compiled, not run on hardware
//
#include <hip/hip_runtime.h>
#include <math.h>

typedef __attribute__((ext_vector_type(16))) _Float16 v16h;
typedef __attribute__((ext_vector_type(2)))  _Float16 h2v;
typedef __attribute__((ext_vector_type(2)))  __fp16   fp16x2;
typedef __attribute__((ext_vector_type(8)))  float    v8f;
typedef __attribute__((ext_vector_type(4)))  unsigned int u32x4;
typedef __attribute__((ext_vector_type(8)))  int      i32x8;
typedef __attribute__((ext_vector_type(4)))  int      i32x4;

#define B_  32
#define V_  512
#define Q_  1024
#define HK  768
#define HD  256
#define HO  2
#define VD  128

// fragment viewed as 16 halfs / 8 packed pairs (_Float16 or __fp16) / 2 dwordx4
union Frag {
  v16h   v;
  h2v    p[8];
  fp16x2 c[8];
  uint4  q[2];
};

__device__ __forceinline__ unsigned short f2hbits(float f) {
  union { _Float16 h; unsigned short s; } u; u.h = (_Float16)f; return u.s;
}
__device__ __forceinline__ float hbits2f(unsigned short s) {
  union { unsigned short s; _Float16 h; } u; u.s = s; return (float)u.h;
}

__device__ __forceinline__ void atomicMaxF(float* addr, float val) {
  if (val >= 0.f) atomicMax((int*)addr, __float_as_int(val));
  else            atomicMin((unsigned int*)addr, __float_as_uint(val));
}

// ---------- init: zero fusion accum, -inf maxes, zero sums, f16 copy of h_mat ----------
__global__ void init_kernel(float* fusion, float* maxbuf, float* sumbuf,
                            const float* __restrict__ h_mat, unsigned short* hm16) {
  int i = blockIdx.x * blockDim.x + threadIdx.x;
  if (i < B_ * HK) fusion[i] = 0.f;
  if (i < B_ * HO) { maxbuf[i] = -__builtin_inff(); sumbuf[i] = 0.f; }
  if (i < HO * HK) hm16[i] = f2hbits(h_mat[i]);
}

// ---------- K1: P = relu(X @ W^T + b) -> f16 (+ optional transposed copy) ----------
__global__ void proj_kernel(const float* __restrict__ X, const float* __restrict__ W,
                            const float* __restrict__ bias,
                            unsigned short* __restrict__ P, unsigned short* __restrict__ PT,
                            int R, int writePT) {
  const int lane = threadIdx.x & 31;
  const int wid  = blockIdx.x * (blockDim.x >> 5) + (threadIdx.x >> 5);
  const int nstrips = HK / 64;                 // 12
  const int mtile  = wid / nstrips;
  const int nstrip = wid - mtile * nstrips;
  if (mtile * 16 >= R) return;
  const int rbase = mtile * 16, nbase = nstrip * 64;
  const int row = lane & 15, hi = lane >> 4;
  const int kbA = hi * 8, kbB = hi * 8;

  v8f acc[4] = {};
  const float* xr = X + (size_t)(rbase + row) * VD;

  for (int kk = 0; kk < VD; kk += 32) {
    Frag a;
    {
      float4 t0 = *(const float4*)(xr + kk + kbA);
      float4 t1 = *(const float4*)(xr + kk + kbA + 4);
      float4 t2 = *(const float4*)(xr + kk + 16 + kbA);
      float4 t3 = *(const float4*)(xr + kk + 16 + kbA + 4);
      a.c[0] = __builtin_amdgcn_cvt_pkrtz(t0.x, t0.y);
      a.c[1] = __builtin_amdgcn_cvt_pkrtz(t0.z, t0.w);
      a.c[2] = __builtin_amdgcn_cvt_pkrtz(t1.x, t1.y);
      a.c[3] = __builtin_amdgcn_cvt_pkrtz(t1.z, t1.w);
      a.c[4] = __builtin_amdgcn_cvt_pkrtz(t2.x, t2.y);
      a.c[5] = __builtin_amdgcn_cvt_pkrtz(t2.z, t2.w);
      a.c[6] = __builtin_amdgcn_cvt_pkrtz(t3.x, t3.y);
      a.c[7] = __builtin_amdgcn_cvt_pkrtz(t3.z, t3.w);
    }
#pragma unroll
    for (int n = 0; n < 4; ++n) {
      const float* wr = W + (size_t)(nbase + n * 16 + row) * VD + kk + kbB;
      Frag bf;
      float4 s0 = *(const float4*)(wr);
      float4 s1 = *(const float4*)(wr + 4);
      float4 s2 = *(const float4*)(wr + 16);
      float4 s3 = *(const float4*)(wr + 20);
      bf.c[0] = __builtin_amdgcn_cvt_pkrtz(s0.x, s0.y);
      bf.c[1] = __builtin_amdgcn_cvt_pkrtz(s0.z, s0.w);
      bf.c[2] = __builtin_amdgcn_cvt_pkrtz(s1.x, s1.y);
      bf.c[3] = __builtin_amdgcn_cvt_pkrtz(s1.z, s1.w);
      bf.c[4] = __builtin_amdgcn_cvt_pkrtz(s2.x, s2.y);
      bf.c[5] = __builtin_amdgcn_cvt_pkrtz(s2.z, s2.w);
      bf.c[6] = __builtin_amdgcn_cvt_pkrtz(s3.x, s3.y);
      bf.c[7] = __builtin_amdgcn_cvt_pkrtz(s3.z, s3.w);
      acc[n] = __builtin_amdgcn_wmma_f32_16x16x32_f16(false, a.v, false, bf.v,
                                                      (short)0, acc[n], false, false);
    }
  }

#pragma unroll
  for (int n = 0; n < 4; ++n) {
    const int ncol = nbase + n * 16 + row;
    const float bv = bias[ncol];
    unsigned short ob[8];
#pragma unroll
    for (int i = 0; i < 8; ++i) {
      float val = acc[n][i] + bv;
      val = val > 0.f ? val : 0.f;
      ob[i] = f2hbits(val);
      P[(size_t)(rbase + hi * 8 + i) * HK + ncol] = ob[i];
    }
    if (writePT) {                       // contiguous 16B transposed store
      uint4 pk;
      pk.x = (unsigned)ob[0] | ((unsigned)ob[1] << 16);
      pk.y = (unsigned)ob[2] | ((unsigned)ob[3] << 16);
      pk.z = (unsigned)ob[4] | ((unsigned)ob[5] << 16);
      pk.w = (unsigned)ob[6] | ((unsigned)ob[7] << 16);
      *(uint4*)(PT + (size_t)ncol * R + rbase + hi * 8) = pk;
    }
  }
}

// ---------- K2: att[b,h] = (v_proj ∘ h_mat[h]) @ q_proj^T + h_bias; per-map max ----------
// Workgroup = 128(v) x 64(q) block tile. B tile (64 rows x 32 halfs) staged into LDS by
// the Tensor Data Mover once per K-step and shared by all 8 waves (ds_load_b128 reads).
__global__ void att_kernel(const unsigned short* __restrict__ vp,
                           const unsigned short* __restrict__ qp,
                           const unsigned short* __restrict__ hm16,
                           const float* __restrict__ h_bias,
                           float* __restrict__ att, float* __restrict__ maxbuf) {
  __shared__ unsigned short btile[64 * 32];          // 4 KB B tile
  const int tid  = threadIdx.x;
  const int lane = tid & 31;
  const int wave = tid >> 5;
  const int blk  = blockIdx.x;
  const int qstrip = blk & 15;            // Q/64
  const int vblk   = (blk >> 4) & 3;      // V/128
  const int h      = (blk >> 6) & 1;
  const int b      = blk >> 7;
  const int rbase = vblk * 128 + wave * 16;
  const int qbase = qstrip * 64;
  const int row = lane & 15, hi = lane >> 4;
  const int kbA = hi * 8, kbB = hi * 8;

  const unsigned short* vrow = vp + ((size_t)b * V_ + rbase + row) * HK;
  const unsigned short* hmr  = hm16 + (size_t)h * HK;
  const unsigned lds_base = (unsigned)(size_t)&btile[0];           // LDS byte offset
  const unsigned long long gbase =
      (unsigned long long)(size_t)qp + 2ull * (((size_t)b * Q_ + qbase) * HK);
  v8f acc[4] = {};

  for (int kk = 0; kk < HK; kk += 32) {
    __syncthreads();                      // previous tile fully consumed
    if (tid < 32) {                       // one wave drives the TDM
      unsigned long long ga = gbase + 2ull * kk;
      u32x4 g0;
      g0[0] = 1u;                                            // count=1, user mode
      g0[1] = lds_base;                                      // lds_addr (bytes)
      g0[2] = (unsigned)(ga & 0xffffffffull);                // global_addr[31:0]
      g0[3] = (unsigned)((ga >> 32) & 0x1ffffffull) | (2u << 30);  // addr[56:32]|type=2
      i32x8 g1;
      g1[0] = 0x10000;                    // data_size = 2 bytes
      g1[1] = (int)(768u << 16);          // tensor_dim0 = 768 (bits 79:48, low half)
      g1[2] = (int)0x80000000u;           // tensor_dim1 = 32768 (bits 111:80, low half)
      g1[3] = (int)(32u << 16);           // tile_dim0 = 32 (bits 127:112)
      g1[4] = 64;                         // tile_dim1 = 64
      g1[5] = 768;                        // tensor_dim0_stride = 768 (bits 207:160)
      g1[6] = 0;
      g1[7] = 0;
      i32x4 gz4 = {0, 0, 0, 0};
      i32x8 gz8 = {0, 0, 0, 0, 0, 0, 0, 0};
      __builtin_amdgcn_tensor_load_to_lds(g0, g1, gz4, gz4, gz8, 0);
      __builtin_amdgcn_s_wait_tensorcnt(0);
    }
    __syncthreads();                      // tile visible to all waves

    __builtin_prefetch(vrow + kk + 32, 0, 3);
    Frag a, hm, as;
    a.q[0]  = *(const uint4*)(vrow + kk + kbA);
    a.q[1]  = *(const uint4*)(vrow + kk + 16 + kbA);
    hm.q[0] = *(const uint4*)(hmr + kk + kbA);
    hm.q[1] = *(const uint4*)(hmr + kk + 16 + kbA);
#pragma unroll
    for (int j = 0; j < 8; ++j) as.p[j] = a.p[j] * hm.p[j];   // v_pk_mul_f16
#pragma unroll
    for (int n = 0; n < 4; ++n) {
      const unsigned short* qrow = &btile[(n * 16 + row) * 32 + kbB];
      Frag bf;
      bf.q[0] = *(const uint4*)(qrow);                        // ds_load_b128
      bf.q[1] = *(const uint4*)(qrow + 16);
      acc[n] = __builtin_amdgcn_wmma_f32_16x16x32_f16(false, as.v, false, bf.v,
                                                      (short)0, acc[n], false, false);
    }
  }

  const float hb = h_bias[h];
  float mx = -__builtin_inff();
  float* abase = att + (size_t)(b * HO + h) * V_ * Q_;
#pragma unroll
  for (int n = 0; n < 4; ++n) {
    const int qcol = qbase + n * 16 + row;
#pragma unroll
    for (int i = 0; i < 8; ++i) {
      float val = acc[n][i] + hb;
      mx = fmaxf(mx, val);
      abase[(size_t)(rbase + hi * 8 + i) * Q_ + qcol] = val;
    }
  }
#pragma unroll
  for (int off = 16; off > 0; off >>= 1) mx = fmaxf(mx, __shfl_xor(mx, off, 32));
  if (lane == 0) atomicMaxF(maxbuf + b * HO + h, mx);
}

// ---------- K3: exp(x - max) in place + per-map sum ----------
__global__ void exp_kernel(float* __restrict__ att, const float* __restrict__ maxbuf,
                           float* __restrict__ sumbuf, const int* __restrict__ smf) {
  if (*smf == 0) return;
  const int map = blockIdx.x >> 9, chunk = blockIdx.x & 511;
  size_t base = (size_t)map * (V_ * Q_) + (size_t)chunk * 1024 + threadIdx.x * 4;
  float4 d = *(float4*)(att + base);
  const float mx = maxbuf[map];
  d.x = __expf(d.x - mx); d.y = __expf(d.y - mx);
  d.z = __expf(d.z - mx); d.w = __expf(d.w - mx);
  *(float4*)(att + base) = d;
  float s = d.x + d.y + d.z + d.w;
#pragma unroll
  for (int off = 16; off > 0; off >>= 1) s += __shfl_xor(s, off, 32);
  __shared__ float red[8];
  if ((threadIdx.x & 31) == 0) red[threadIdx.x >> 5] = s;
  __syncthreads();
  if (threadIdx.x == 0) {
    float t = 0.f;
#pragma unroll
    for (int i = 0; i < 8; ++i) t += red[i];
    atomicAdd(sumbuf + map, t);
  }
}

// ---------- K4: normalize by per-map sum ----------
__global__ void norm_kernel(float* __restrict__ att, const float* __restrict__ sumbuf,
                            const int* __restrict__ smf) {
  if (*smf == 0) return;
  const int map = blockIdx.x >> 9, chunk = blockIdx.x & 511;
  size_t base = (size_t)map * (V_ * Q_) + (size_t)chunk * 1024 + threadIdx.x * 4;
  const float inv = 1.0f / sumbuf[map];
  float4 d = *(float4*)(att + base);
  d.x *= inv; d.y *= inv; d.z *= inv; d.w *= inv;
  *(float4*)(att + base) = d;
}

// ---------- K5: t = att @ q_projT, fusion[b,k] += Σ_v v_proj[v,k]·t[v,k] ----------
__global__ void fusion_kernel(const float* __restrict__ att,
                              const unsigned short* __restrict__ vp,
                              const unsigned short* __restrict__ qT,
                              float* __restrict__ fusion) {
  const int lane = threadIdx.x & 31;
  const int wid  = blockIdx.x * (blockDim.x >> 5) + (threadIdx.x >> 5);
  const int kstrip = wid % 12;
  const int vt = (wid / 12) & 31;
  const int h  = (wid / 384) & 1;
  const int b  = wid / 768;
  const int rbase = vt * 16;
  const int row = lane & 15, hi = lane >> 4;
  const int kbA = hi * 8, kbB = hi * 8;

  const float* arow = att + ((size_t)(b * HO + h) * V_ + rbase + row) * Q_;
  v8f acc[4] = {};

  for (int qq = 0; qq < Q_; qq += 32) {
    __builtin_prefetch(arow + qq + 32, 0, 3);
    Frag a;
    {
      float4 t0 = *(const float4*)(arow + qq + kbA);
      float4 t1 = *(const float4*)(arow + qq + kbA + 4);
      float4 t2 = *(const float4*)(arow + qq + 16 + kbA);
      float4 t3 = *(const float4*)(arow + qq + 16 + kbA + 4);
      a.c[0] = __builtin_amdgcn_cvt_pkrtz(t0.x, t0.y);
      a.c[1] = __builtin_amdgcn_cvt_pkrtz(t0.z, t0.w);
      a.c[2] = __builtin_amdgcn_cvt_pkrtz(t1.x, t1.y);
      a.c[3] = __builtin_amdgcn_cvt_pkrtz(t1.z, t1.w);
      a.c[4] = __builtin_amdgcn_cvt_pkrtz(t2.x, t2.y);
      a.c[5] = __builtin_amdgcn_cvt_pkrtz(t2.z, t2.w);
      a.c[6] = __builtin_amdgcn_cvt_pkrtz(t3.x, t3.y);
      a.c[7] = __builtin_amdgcn_cvt_pkrtz(t3.z, t3.w);
    }
#pragma unroll
    for (int n = 0; n < 4; ++n) {
      const unsigned short* qrow =
          qT + (size_t)(kstrip * 64 + n * 16 + row) * (B_ * Q_) + (size_t)b * Q_ + qq + kbB;
      Frag bf;
      bf.q[0] = *(const uint4*)(qrow);
      bf.q[1] = *(const uint4*)(qrow + 16);
      acc[n] = __builtin_amdgcn_wmma_f32_16x16x32_f16(false, a.v, false, bf.v,
                                                      (short)0, acc[n], false, false);
    }
  }

#pragma unroll
  for (int n = 0; n < 4; ++n) {
    const int kn = kstrip * 64 + n * 16 + row;
    float p = 0.f;
#pragma unroll
    for (int i = 0; i < 8; ++i) {
      float vv = hbits2f(vp[((size_t)b * V_ + rbase + hi * 8 + i) * HK + kn]);
      p += acc[n][i] * vv;
    }
    p += __shfl_xor(p, 16, 32);
    if (lane < 16) atomicAdd(fusion + b * HK + kn, p);
  }
}

// ---------- K6: k=3 group-sum + BatchNorm over batch ----------
__global__ void bn_kernel(const float* __restrict__ fusion,
                          const float* __restrict__ gamma, const float* __restrict__ beta,
                          float* __restrict__ out) {
  const int c = threadIdx.x;
  float vals[B_];
  float mean = 0.f;
  for (int b = 0; b < B_; ++b) {
    const float* f = fusion + (size_t)b * HK + c * 3;
    float s = f[0] + f[1] + f[2];
    vals[b] = s; mean += s;
  }
  mean *= (1.0f / B_);
  float var = 0.f;
  for (int b = 0; b < B_; ++b) { float d = vals[b] - mean; var += d * d; }
  var *= (1.0f / B_);
  const float inv = rsqrtf(var + 1e-5f) * gamma[c];
  const float bt = beta[c];
  for (int b = 0; b < B_; ++b)
    out[(size_t)b * HD + c] = (vals[b] - mean) * inv + bt;
}

extern "C" void kernel_launch(void* const* d_in, const int* in_sizes, int n_in,
                              void* d_out, int out_size, void* d_ws, size_t ws_size,
                              hipStream_t stream) {
  const float* v      = (const float*)d_in[0];
  const float* q      = (const float*)d_in[1];
  const float* Wv     = (const float*)d_in[2];
  const float* bv     = (const float*)d_in[3];
  const float* Wq     = (const float*)d_in[4];
  const float* bq     = (const float*)d_in[5];
  const float* h_mat  = (const float*)d_in[6];
  const float* h_bias = (const float*)d_in[7];
  const float* gamma  = (const float*)d_in[8];
  const float* beta   = (const float*)d_in[9];
  const int*   smf    = (const int*)d_in[10];

  float* out    = (float*)d_out;
  float* logits = out;                 // [32,256]
  float* att    = out + B_ * HD;       // [32,2,512,1024]

  char* ws = (char*)d_ws;
  unsigned short* v_proj  = (unsigned short*)ws;                    // 25,165,824 B
  unsigned short* q_proj  = (unsigned short*)(ws + 25165824);       // 50,331,648 B
  unsigned short* q_projT = (unsigned short*)(ws + 75497472);       // 50,331,648 B
  float* maxbuf = (float*)(ws + 125829120);
  float* sumbuf = (float*)(ws + 125829376);
  float* fusion = (float*)(ws + 125829632);
  unsigned short* hm16 = (unsigned short*)(ws + 125927936);         // 3,072 B

  init_kernel<<<96, 256, 0, stream>>>(fusion, maxbuf, sumbuf, h_mat, hm16);
  proj_kernel<<<1536, 256, 0, stream>>>(v, Wv, bv, v_proj, (unsigned short*)nullptr,
                                        B_ * V_, 0);
  proj_kernel<<<3072, 256, 0, stream>>>(q, Wq, bq, q_proj, q_projT, B_ * Q_, 1);
  att_kernel<<<4096, 256, 0, stream>>>(v_proj, q_proj, hm16, h_bias, att, maxbuf);
  exp_kernel<<<32768, 256, 0, stream>>>(att, maxbuf, sumbuf, smf);
  norm_kernel<<<32768, 256, 0, stream>>>(att, sumbuf, smf);
  fusion_kernel<<<3072, 256, 0, stream>>>(att, v_proj, q_projT, fusion);
  bn_kernel<<<1, 256, 0, stream>>>(fusion, gamma, beta, logits);
}
